// BiLSTM_CRF_54778012893717
// MI455X (gfx1250) — matrix operand order repaired and checked by reading, not compiled, not run on hardware
//
#include <hip/hip_runtime.h>

// ---------------------------------------------------------------------------
// Types for CDNA5 WMMA (wave32): D(16x16 f32) = A(16x32 f16) x B(32x16 f16) + C
// ---------------------------------------------------------------------------
typedef _Float16 v16h __attribute__((ext_vector_type(16)));
typedef _Float16 h8   __attribute__((ext_vector_type(8)));
typedef float    v8f  __attribute__((ext_vector_type(8)));

#define S_LEN   128
#define BATCH   32
#define WLEN    16
#define EDIM    300
#define CEDIM   32
#define CONVOUT 128
#define DIN1    428
#define DIN1P   448
#define HID     512
#define G4H     2048
#define EMB2    1024
#define QKVN    3072
#define NTAGS   20
#define ROWS    (S_LEN*BATCH)   // 4096 token rows, row = s*B + b

__device__ __forceinline__ v8f wmma16x16x32(v16h a, v16h b, v8f c) {
  return __builtin_amdgcn_wmma_f32_16x16x32_f16(false, a, false, b, (short)0, c,
                                                false, false);
}

// Branch-free fast transcendentals for the recurrence (v_exp_f32 based).
__device__ __forceinline__ float fast_sigmoid(float x) {
  return 1.f / (1.f + __expf(-x));
}
__device__ __forceinline__ float fast_tanh(float x) {
  return 1.f - 2.f / (__expf(2.f * x) + 1.f);
}

// A fragment: 16x32 f16 tile from row-major [M, lda] activations.
// ISA layout: lane<16 halves = K {0..7,16..23}; lane>=16 = K {8..15,24..31}; M = lane&15.
__device__ __forceinline__ v16h load_a_tile(const _Float16* A, int lda,
                                            int mbase, int kbase, int lane) {
  int m   = mbase + (lane & 15);
  int off = (lane >= 16) ? 8 : 0;
  const _Float16* p = A + (size_t)m * lda + kbase + off;
  union { v16h v; h8 h[2]; } u;
  u.h[0] = *(const h8*)(p);
  u.h[1] = *(const h8*)(p + 16);
  return u.v;
}

// ---------------------------------------------------------------------------
// Pack fp32 weight W[N,K] (row-major, out x in) into WMMA B tiles:
// dst[((nt*ktiles)+kt)*512 + l*16 + kp] = W[nt*16 + (l&15)][kt*32 + (kp>>3)*16 + (l>>4)*8 + (kp&7)]
// (zero-padded past N,K)
// ---------------------------------------------------------------------------
__global__ void pack_b_kernel(const float* __restrict__ W, _Float16* __restrict__ dst,
                              int N, int K, int ktiles, int ntiles) {
  size_t total = (size_t)ntiles * ktiles * 512;
  for (size_t idx = (size_t)blockIdx.x * blockDim.x + threadIdx.x; idx < total;
       idx += (size_t)gridDim.x * blockDim.x) {
    int    r    = (int)(idx & 511);
    size_t tile = idx >> 9;
    int    kt   = (int)(tile % ktiles);
    int    nt   = (int)(tile / ktiles);
    int l = r >> 4, kp = r & 15;
    int n = nt * 16 + (l & 15);
    int k = kt * 32 + ((kp >> 3) << 4) + ((l >> 4) << 3) + (kp & 7);
    float v = (n < N && k < K) ? W[(size_t)n * K + k] : 0.f;
    dst[idx] = (_Float16)v;
  }
}

// ---------------------------------------------------------------------------
// Generic WMMA GEMM: C[M,N] = A[M,Kpad] @ packedB + bias1 + bias2
// one 16x16 output tile per wave; 8 waves per block.
// ---------------------------------------------------------------------------
template <bool OUT_HALF>
__global__ __launch_bounds__(256) void gemm_wmma_kernel(
    const _Float16* __restrict__ A, int lda,
    const _Float16* __restrict__ Bp,
    const float* __restrict__ bias1, const float* __restrict__ bias2,
    void* __restrict__ Cout, int ldc, int M, int N, int ktiles, int ntiles) {
  int wave = blockIdx.x * (blockDim.x >> 5) + (threadIdx.x >> 5);
  int lane = threadIdx.x & 31;
  int total = (M >> 4) * ntiles;
  if (wave >= total) return;
  int mt = wave / ntiles;
  int nt = wave % ntiles;

  v8f acc = {0.f, 0.f, 0.f, 0.f, 0.f, 0.f, 0.f, 0.f};
  const _Float16* bptr = Bp + ((size_t)nt * ktiles) * 512 + lane * 16;
  for (int kt = 0; kt < ktiles; ++kt) {
    v16h a = load_a_tile(A, lda, mt * 16, kt * 32, lane);
    v16h b = *(const v16h*)(bptr);
    bptr += 512;
    acc = wmma16x16x32(a, b, acc);
  }
  int n = nt * 16 + (lane & 15);
  float bsum = 0.f;
  if (n < N) {
    if (bias1) bsum += bias1[n];
    if (bias2) bsum += bias2[n];
  }
  int m0 = mt * 16 + ((lane >> 4) << 3);
#pragma unroll
  for (int r = 0; r < 8; ++r) {
    int m = m0 + r;
    if (m < M && n < N) {
      float v = acc[r] + bsum;
      if (OUT_HALF) ((_Float16*)Cout)[(size_t)m * ldc + n] = (_Float16)v;
      else          ((float*)Cout)[(size_t)m * ldc + n]    = v;
    }
  }
}

// ---------------------------------------------------------------------------
// Word embedding -> X1[:, 0:300], zero pad [428:448)
// ---------------------------------------------------------------------------
__global__ void word_embed_kernel(const int* __restrict__ words,
                                  const float* __restrict__ wemb,
                                  _Float16* __restrict__ X1) {
  int row = blockIdx.x;
  int wid = words[row];
  const float* src = wemb + (size_t)wid * EDIM;
  for (int e = threadIdx.x; e < EDIM; e += blockDim.x)
    X1[(size_t)row * DIN1P + e] = (_Float16)src[e];
  for (int e = DIN1 + threadIdx.x; e < DIN1P; e += blockDim.x)
    X1[(size_t)row * DIN1P + e] = (_Float16)0.f;
}

// ---------------------------------------------------------------------------
// Grouped char conv (K=3, groups=32, 4 filters/group) + max pool -> X1[:,300:428)
// ---------------------------------------------------------------------------
__global__ void char_conv_kernel(const int* __restrict__ chars,
                                 const float* __restrict__ cemb_tab,
                                 const float* __restrict__ cw,
                                 const float* __restrict__ cb,
                                 _Float16* __restrict__ X1) {
  int row = blockIdx.x;           // row = s*B + b
  int s = row >> 5, b = row & 31;
  __shared__ float cemb[WLEN][CEDIM];
  const int* cp = chars + ((size_t)b * S_LEN + s) * WLEN;
  for (int i = threadIdx.x; i < WLEN * CEDIM; i += blockDim.x) {
    int w_ = i >> 5, c = i & 31;
    cemb[w_][c] = cemb_tab[(size_t)cp[w_] * CEDIM + c];
  }
  __syncthreads();
  int oc = threadIdx.x;           // 128 output channels
  int c  = oc >> 2;               // input channel of this group
  float w0 = cw[oc * 3 + 0], w1 = cw[oc * 3 + 1], w2 = cw[oc * 3 + 2];
  float mx = -3.4e38f;
#pragma unroll
  for (int p = 0; p < WLEN - 2; ++p) {
    float v = w0 * cemb[p][c] + w1 * cemb[p + 1][c] + w2 * cemb[p + 2][c];
    mx = fmaxf(mx, v);
  }
  X1[(size_t)row * DIN1P + EDIM + oc] = (_Float16)(mx + cb[oc]);
}

// ---------------------------------------------------------------------------
// Sequential BiLSTM recurrence. grid=2 (dir), block=1024 (32 waves), dyn LDS:
// double-buffered h[32][512] fp16. Each wave owns 2 (mt,nt) units; per step each
// unit does 4 gates x 16 K-tiles of v_wmma_f32_16x16x32_f16 against packed Whh.
// Gpre = x@Wih^T + bih + bhh is precomputed by the big GEMM.
//
// NOTE: an opaque SGPR *offset* (not pointer) defeats LICM of the step-invariant
// weight loads without destroying global address-space inference: B tiles stay
// in-loop global_load_b128 (L2-resident), no scratch spill, no FLAT path.
// ---------------------------------------------------------------------------
extern __shared__ char g_smem[];

__global__ __launch_bounds__(1024) void lstm_seq_kernel(
    const float* __restrict__ GpreF, const float* __restrict__ GpreB,
    const _Float16* __restrict__ WhhPF, const _Float16* __restrict__ WhhPB,
    _Float16* __restrict__ Xout) {
  const int dir = blockIdx.x;
  const float*    Gpre = dir ? GpreB : GpreF;
  const _Float16* Whh  = dir ? WhhPB : WhhPF;
  _Float16* hbuf = (_Float16*)g_smem;                 // [2][32*512]
  const int tid = threadIdx.x, lane = tid & 31, wave = tid >> 5;

  for (int i = tid; i < BATCH * HID; i += 1024) hbuf[i] = (_Float16)0.f;
  __syncthreads();

  float creg[2][8];
#pragma unroll
  for (int u = 0; u < 2; ++u)
#pragma unroll
    for (int r = 0; r < 8; ++r) creg[u][r] = 0.f;

#pragma clang loop unroll(disable)
  for (int step = 0; step < S_LEN; ++step) {
    int t = dir ? (S_LEN - 1 - step) : step;
    int cur = step & 1, nxt = cur ^ 1;
    const _Float16* hcur = hbuf + cur * (BATCH * HID);
    _Float16*       hnxt = hbuf + nxt * (BATCH * HID);

    // Defeat LICM: opaque (but zero) scalar offset recomputed per timestep so
    // the B-tile loads cannot be hoisted+spilled; base pointer keeps its
    // global address space => global_load_b128, LOADcnt only.
    unsigned int wofs = 0;
    asm volatile("" : "+s"(wofs));
    const _Float16* WhhS = Whh + wofs;

#pragma unroll
    for (int u = 0; u < 2; ++u) {
      int unit = wave * 2 + u;
      int mt = unit >> 5;          // 0..1  (batch tile)
      int nt = unit & 31;          // 0..31 (hidden tile)
      v8f a0 = {0.f,0.f,0.f,0.f,0.f,0.f,0.f,0.f};
      v8f a1 = a0, a2 = a0, a3 = a0;
#pragma clang loop unroll(disable)
      for (int kt = 0; kt < HID / 32; ++kt) {
        v16h a = load_a_tile(hcur, HID, mt * 16, kt * 32, lane);
        const _Float16* base = WhhS + lane * 16;
        v16h b0 = *(const v16h*)(base + ((size_t)((0 * 32 + nt) * 16 + kt)) * 512);
        v16h b1 = *(const v16h*)(base + ((size_t)((1 * 32 + nt) * 16 + kt)) * 512);
        v16h b2 = *(const v16h*)(base + ((size_t)((2 * 32 + nt) * 16 + kt)) * 512);
        v16h b3 = *(const v16h*)(base + ((size_t)((3 * 32 + nt) * 16 + kt)) * 512);
        a0 = wmma16x16x32(a, b0, a0);
        a1 = wmma16x16x32(a, b1, a1);
        a2 = wmma16x16x32(a, b2, a2);
        a3 = wmma16x16x32(a, b3, a3);
      }
      int n  = nt * 16 + (lane & 15);
      int m0 = mt * 16 + ((lane >> 4) << 3);
#pragma unroll
      for (int r = 0; r < 8; ++r) {
        int m = m0 + r;                                     // batch index
        size_t grow = ((size_t)t * BATCH + m) * G4H;
        float gi = a0[r] + Gpre[grow + 0 * HID + n];
        float gf = a1[r] + Gpre[grow + 1 * HID + n];
        float gg = a2[r] + Gpre[grow + 2 * HID + n];
        float go = a3[r] + Gpre[grow + 3 * HID + n];
        float is = fast_sigmoid(gi);
        float fs = fast_sigmoid(gf);
        float gt = fast_tanh(gg);
        float os = fast_sigmoid(go);
        float cc = fs * creg[u][r] + is * gt;
        creg[u][r] = cc;
        float hh = os * fast_tanh(cc);
        hnxt[m * HID + n] = (_Float16)hh;
        Xout[((size_t)t * BATCH + m) * EMB2 + dir * HID + n] = (_Float16)hh;
      }
    }
    __syncthreads();
  }
}

// ---------------------------------------------------------------------------
// Self-attention (8 heads, Dh=128). grid=B, block=128 (thread=query row).
// Dynamic LDS: Qs/Ks/Vs fp16 (32KB each) + P f32 (64KB) + mask (512B).
// Writes merged fp16 context and head-averaged attention weights into d_out.
// ---------------------------------------------------------------------------
__global__ __launch_bounds__(128) void attention_kernel(
    const _Float16* __restrict__ QKV,     // [ROWS, 3072] fp16
    const int* __restrict__ words,
    _Float16* __restrict__ ctx,           // [ROWS, 1024] fp16
    float* __restrict__ attn_out) {       // [B, S, S] f32 (in d_out)
  const int b = blockIdx.x;
  const int q = threadIdx.x;              // 0..127
  _Float16* Qs = (_Float16*)g_smem;       // 128*128
  _Float16* Ks = Qs + S_LEN * 128;
  _Float16* Vs = Ks + S_LEN * 128;
  float*    P  = (float*)(Vs + S_LEN * 128);   // 128*128
  float*    mk = P + S_LEN * S_LEN;            // 128
  const float scale = 0.0883883476483184f;     // 1/sqrt(128)

  mk[q] = (words[q * BATCH + b] == 0) ? -1e9f : 0.f;

  for (int h = 0; h < 8; ++h) {
    __syncthreads();
    for (int i = threadIdx.x; i < S_LEN * 128; i += 128) {
      int s_ = i >> 7, d = i & 127;
      size_t rr = ((size_t)s_ * BATCH + b) * QKVN;
      Qs[i] = QKV[rr + 0    + h * 128 + d];
      Ks[i] = QKV[rr + 1024 + h * 128 + d];
      Vs[i] = QKV[rr + 2048 + h * 128 + d];
    }
    __syncthreads();

    // scores + row max
    float mx = -3.4e38f;
    for (int k = 0; k < S_LEN; ++k) {
      float s = 0.f;
      const _Float16* qr = &Qs[q * 128];
      const _Float16* kr = &Ks[k * 128];
      for (int d = 0; d < 128; ++d) s += (float)qr[d] * (float)kr[d];
      s = s * scale + mk[k];
      P[q * S_LEN + k] = s;
      mx = fmaxf(mx, s);
    }
    // softmax row; store probs; accumulate averaged attention
    float sum = 0.f;
    for (int k = 0; k < S_LEN; ++k) sum += __expf(P[q * S_LEN + k] - mx);
    float inv = 1.f / sum;
    float* aw = attn_out + ((size_t)b * S_LEN + q) * S_LEN;
    for (int k = 0; k < S_LEN; ++k) {
      float p = __expf(P[q * S_LEN + k] - mx) * inv;
      P[q * S_LEN + k] = p;
      float contrib = p * 0.125f;
      if (h == 0) aw[k] = contrib; else aw[k] += contrib;
    }
    __syncthreads();
    // context = P @ V
    size_t rr = ((size_t)q * BATCH + b) * EMB2 + h * 128;
    for (int d = 0; d < 128; ++d) {
      float o = 0.f;
      for (int k = 0; k < S_LEN; ++k) o += P[q * S_LEN + k] * (float)Vs[k * 128 + d];
      ctx[rr + d] = (_Float16)o;
    }
  }
}

// ---------------------------------------------------------------------------
// CRF negative log-likelihood (reduction='sum'). One block, 32 batches x 32 lanes.
// ---------------------------------------------------------------------------
__global__ __launch_bounds__(1024) void crf_llh_kernel(
    const float* __restrict__ emis,   // [ROWS, 32] (cols 0..19 valid)
    const int* __restrict__ tags, const int* __restrict__ words,
    const float* __restrict__ start, const float* __restrict__ endv,
    const float* __restrict__ trans, float* __restrict__ loss_out) {
  int b = threadIdx.x >> 5, j = threadIdx.x & 31;
  __shared__ float sc[2][BATCH][32];
  __shared__ float llh[BATCH];
  sc[0][b][j] = (j < NTAGS) ? (start[j] + emis[(size_t)b * 32 + j]) : -1e30f;
  __syncthreads();
  for (int s = 1; s < S_LEN; ++s) {
    int cur = (s - 1) & 1, nxt = s & 1;
    bool m = words[s * BATCH + b] != 0;
    float v = -1e30f;
    if (j < NTAGS) {
      float mx = -3.4e38f;
      for (int i = 0; i < NTAGS; ++i) mx = fmaxf(mx, sc[cur][b][i] + trans[i * NTAGS + j]);
      float sum = 0.f;
      for (int i = 0; i < NTAGS; ++i) sum += __expf(sc[cur][b][i] + trans[i * NTAGS + j] - mx);
      float nv = mx + __logf(sum) + emis[((size_t)s * BATCH + b) * 32 + j];
      v = m ? nv : sc[cur][b][j];
    }
    sc[nxt][b][j] = v;
    __syncthreads();
  }
  if (j == 0) {
    float mx = -3.4e38f;
    for (int i = 0; i < NTAGS; ++i) mx = fmaxf(mx, sc[1][b][i] + endv[i]);
    float sum = 0.f;
    for (int i = 0; i < NTAGS; ++i) sum += __expf(sc[1][b][i] + endv[i] - mx);
    float den = mx + __logf(sum);
    int prev = tags[b];
    float num = start[prev] + emis[(size_t)b * 32 + prev];
    for (int s = 1; s < S_LEN; ++s) {
      if (words[s * BATCH + b] != 0) {
        int tc = tags[s * BATCH + b];
        num += trans[prev * NTAGS + tc] + emis[((size_t)s * BATCH + b) * 32 + tc];
        prev = tc;
      }
    }
    num += endv[prev];
    llh[b] = num - den;
  }
  __syncthreads();
  if (threadIdx.x == 0) {
    float s = 0.f;
    for (int i = 0; i < BATCH; ++i) s += llh[i];
    *loss_out = -s;
  }
}

// ---------------------------------------------------------------------------
// Viterbi decode. grid=B, block=32 (thread=tag). Tags written as floats.
// ---------------------------------------------------------------------------
__global__ __launch_bounds__(32) void viterbi_kernel(
    const float* __restrict__ emis, const int* __restrict__ words,
    const float* __restrict__ start, const float* __restrict__ endv,
    const float* __restrict__ trans, float* __restrict__ out_tags) {
  int b = blockIdx.x, j = threadIdx.x;
  __shared__ float sc[2][32];
  __shared__ int   hist[S_LEN * NTAGS];
  if (j < NTAGS) sc[0][j] = start[j] + emis[(size_t)b * 32 + j];
  __syncthreads();
  for (int s = 1; s < S_LEN; ++s) {
    int cur = (s - 1) & 1, nxt = s & 1;
    bool m = words[s * BATCH + b] != 0;
    if (j < NTAGS) {
      float mx = -3.4e38f; int bi = 0;
      for (int i = 0; i < NTAGS; ++i) {
        float v = sc[cur][i] + trans[i * NTAGS + j];
        if (v > mx) { mx = v; bi = i; }
      }
      float nv = mx + emis[((size_t)s * BATCH + b) * 32 + j];
      sc[nxt][j] = m ? nv : sc[cur][j];
      hist[s * NTAGS + j] = m ? bi : j;
    }
    __syncthreads();
  }
  if (j == 0) {
    float mx = -3.4e38f; int tag = 0;
    for (int i = 0; i < NTAGS; ++i) {
      float v = sc[1][i] + endv[i];
      if (v > mx) { mx = v; tag = i; }
    }
    for (int s = S_LEN - 1; s >= 1; --s) {
      out_tags[s * BATCH + b] = (float)tag;
      tag = hist[s * NTAGS + tag];
    }
    out_tags[b] = (float)tag;
  }
}

// ---------------------------------------------------------------------------
// Host: orchestrate the pipeline on `stream` (graph-capture safe).
// ---------------------------------------------------------------------------
extern "C" void kernel_launch(void* const* d_in, const int* in_sizes, int n_in,
                              void* d_out, int out_size, void* d_ws, size_t ws_size,
                              hipStream_t stream) {
  const int*   words   = (const int*)d_in[0];
  const int*   chars   = (const int*)d_in[1];
  const int*   tags    = (const int*)d_in[2];
  const float* wemb    = (const float*)d_in[3];
  const float* cembT   = (const float*)d_in[4];
  const float* convw   = (const float*)d_in[5];
  const float* convb   = (const float*)d_in[6];
  // lstm layer0: 7..14  (Wih_f,Whh_f,bih_f,bhh_f, Wih_b,Whh_b,bih_b,bhh_b)
  // lstm layer1: 15..22
  const float* Wih1f = (const float*)d_in[7],  *Whh1f = (const float*)d_in[8];
  const float* bih1f = (const float*)d_in[9],  *bhh1f = (const float*)d_in[10];
  const float* Wih1b = (const float*)d_in[11], *Whh1b = (const float*)d_in[12];
  const float* bih1b = (const float*)d_in[13], *bhh1b = (const float*)d_in[14];
  const float* Wih2f = (const float*)d_in[15], *Whh2f = (const float*)d_in[16];
  const float* bih2f = (const float*)d_in[17], *bhh2f = (const float*)d_in[18];
  const float* Wih2b = (const float*)d_in[19], *Whh2b = (const float*)d_in[20];
  const float* bih2b = (const float*)d_in[21], *bhh2b = (const float*)d_in[22];
  const float* Wqkv  = (const float*)d_in[23], *bqkv  = (const float*)d_in[24];
  const float* Wo    = (const float*)d_in[25], *bo    = (const float*)d_in[26];
  const float* fcw   = (const float*)d_in[27], *fcb   = (const float*)d_in[28];
  const float* crfS  = (const float*)d_in[29], *crfE  = (const float*)d_in[30];
  const float* crfT  = (const float*)d_in[31];

  float* out_tags = (float*)d_out;
  float* out_loss = out_tags + ROWS;
  float* out_attn = out_loss + 1;

  // ---- workspace carve ----
  char* wp = (char*)d_ws;
  auto carve = [&](size_t bytes) -> void* {
    void* p = (void*)wp;
    wp += (bytes + 255) & ~(size_t)255;
    return p;
  };
  _Float16* X1    = (_Float16*)carve((size_t)ROWS * DIN1P * 2);
  _Float16* X2    = (_Float16*)carve((size_t)ROWS * EMB2 * 2);
  _Float16* X3    = (_Float16*)carve((size_t)ROWS * EMB2 * 2);
  float*    GpreF = (float*)carve((size_t)ROWS * G4H * 4);
  float*    GpreB = (float*)carve((size_t)ROWS * G4H * 4);
  _Float16* QKVh  = (_Float16*)carve((size_t)ROWS * QKVN * 2);
  _Float16* CTX   = (_Float16*)carve((size_t)ROWS * EMB2 * 2);
  _Float16* PROJ  = (_Float16*)carve((size_t)ROWS * EMB2 * 2);
  float*    EMIS  = (float*)carve((size_t)ROWS * 32 * 4);
  auto packbuf = [&](int ntiles, int ktiles) -> _Float16* {
    return (_Float16*)carve((size_t)ntiles * ktiles * 512 * 2);
  };
  _Float16* pWih1f = packbuf(128, 14), *pWih1b = packbuf(128, 14);
  _Float16* pWhh1f = packbuf(128, 16), *pWhh1b = packbuf(128, 16);
  _Float16* pWih2f = packbuf(128, 32), *pWih2b = packbuf(128, 32);
  _Float16* pWhh2f = packbuf(128, 16), *pWhh2b = packbuf(128, 16);
  _Float16* pWqkv  = packbuf(192, 32);
  _Float16* pWo    = packbuf(64, 32);
  _Float16* pFc    = packbuf(2, 32);

  // ---- pack weights into WMMA B-tile layout ----
  pack_b_kernel<<<256, 256, 0, stream>>>(Wih1f, pWih1f, G4H, DIN1, 14, 128);
  pack_b_kernel<<<256, 256, 0, stream>>>(Wih1b, pWih1b, G4H, DIN1, 14, 128);
  pack_b_kernel<<<256, 256, 0, stream>>>(Whh1f, pWhh1f, G4H, HID, 16, 128);
  pack_b_kernel<<<256, 256, 0, stream>>>(Whh1b, pWhh1b, G4H, HID, 16, 128);
  pack_b_kernel<<<256, 256, 0, stream>>>(Wih2f, pWih2f, G4H, EMB2, 32, 128);
  pack_b_kernel<<<256, 256, 0, stream>>>(Wih2b, pWih2b, G4H, EMB2, 32, 128);
  pack_b_kernel<<<256, 256, 0, stream>>>(Whh2f, pWhh2f, G4H, HID, 16, 128);
  pack_b_kernel<<<256, 256, 0, stream>>>(Whh2b, pWhh2b, G4H, HID, 16, 128);
  pack_b_kernel<<<256, 256, 0, stream>>>(Wqkv,  pWqkv,  QKVN, EMB2, 32, 192);
  pack_b_kernel<<<256, 256, 0, stream>>>(Wo,    pWo,    EMB2, EMB2, 32, 64);
  pack_b_kernel<<<256, 256, 0, stream>>>(fcw,   pFc,    NTAGS, EMB2, 32, 2);

  // ---- embeddings + char CNN -> X1 [ROWS, 448] fp16 ----
  word_embed_kernel<<<ROWS, 128, 0, stream>>>(words, wemb, X1);
  char_conv_kernel<<<ROWS, 128, 0, stream>>>(chars, cembT, convw, convb, X1);

  auto gemm = [&](const _Float16* A, int lda, const _Float16* Bp,
                  const float* b1, const float* b2, void* C, int ldc,
                  int M, int N, int ktiles, int ntiles, bool half) {
    int tiles  = (M / 16) * ntiles;
    int blocks = (tiles + 7) / 8;
    if (half)
      gemm_wmma_kernel<true><<<blocks, 256, 0, stream>>>(A, lda, Bp, b1, b2, C,
                                                         ldc, M, N, ktiles, ntiles);
    else
      gemm_wmma_kernel<false><<<blocks, 256, 0, stream>>>(A, lda, Bp, b1, b2, C,
                                                          ldc, M, N, ktiles, ntiles);
  };

  const size_t lstm_lds = 2 * (size_t)BATCH * HID * sizeof(_Float16);  // 64 KB

  // ---- layer 1: input projection GEMMs, then recurrence ----
  gemm(X1, DIN1P, pWih1f, bih1f, bhh1f, GpreF, G4H, ROWS, G4H, 14, 128, false);
  gemm(X1, DIN1P, pWih1b, bih1b, bhh1b, GpreB, G4H, ROWS, G4H, 14, 128, false);
  lstm_seq_kernel<<<2, 1024, lstm_lds, stream>>>(GpreF, GpreB, pWhh1f, pWhh1b, X2);

  // ---- layer 2 ----
  gemm(X2, EMB2, pWih2f, bih2f, bhh2f, GpreF, G4H, ROWS, G4H, 32, 128, false);
  gemm(X2, EMB2, pWih2b, bih2b, bhh2b, GpreB, G4H, ROWS, G4H, 32, 128, false);
  lstm_seq_kernel<<<2, 1024, lstm_lds, stream>>>(GpreF, GpreB, pWhh2f, pWhh2b, X3);

  // ---- attention ----
  gemm(X3, EMB2, pWqkv, bqkv, nullptr, QKVh, QKVN, ROWS, QKVN, 32, 192, true);
  const size_t attn_lds = (size_t)3 * S_LEN * 128 * sizeof(_Float16) +
                          (size_t)S_LEN * S_LEN * sizeof(float) + S_LEN * sizeof(float);
  attention_kernel<<<BATCH, 128, attn_lds, stream>>>(QKVh, words, CTX, out_attn);
  gemm(CTX, EMB2, pWo, bo, nullptr, PROJ, EMB2, ROWS, EMB2, 32, 64, true);

  // ---- emissions + CRF ----
  gemm(PROJ, EMB2, pFc, fcb, nullptr, EMIS, 32, ROWS, NTAGS, 32, 2, false);
  crf_llh_kernel<<<1, 1024, 0, stream>>>(EMIS, tags, words, crfS, crfE, crfT, out_loss);
  viterbi_kernel<<<BATCH, 32, 0, stream>>>(EMIS, words, crfS, crfE, crfT, out_tags);

  (void)in_sizes; (void)n_in; (void)out_size; (void)ws_size;
}
